// MaskedGridPredictor_48558900249394
// MI455X (gfx1250) — matrix operand order repaired and checked by reading, not compiled, not run on hardware
//
#include <hip/hip_runtime.h>
#include <stdint.h>

#define B_      4
#define N_      8192
#define N2_     8192
#define DIM_    256
#define HEADS_  8
#define DH_     64
#define INNER_  512
#define MTOT    (B_ * N_)          // 32768 rows (b*n flattened)
#define KSPLIT  8
#define POS_SCALE 64.0f            // SCALE / MIN_FREQ
#define LOG2_10000_OVER16 0.83048202372003362f   // log2(10000)/16

typedef __attribute__((ext_vector_type(16))) __bf16 v16bf;
typedef __attribute__((ext_vector_type(8)))  float  v8f;

// ---------------------------------------------------------------------------
// WMMA helpers (CDNA5 wave32, D = A(16x32 bf16) x B(32x16 bf16) + C(16x16 f32))
// ---------------------------------------------------------------------------
__device__ __forceinline__ v8f wmma_bf16(v16bf a, v16bf b, v8f c) {
  return __builtin_amdgcn_wmma_f32_16x16x32_bf16(
      /*neg_a=*/false, a, /*neg_b=*/false, b,
      /*c_mod=*/(short)0, c, /*reuse_a=*/false, /*reuse_b=*/false);
}

// A fragment: 16x32 (MxK), source row-major, K contiguous, row stride ld (elems).
// Lane m = lane&15, lane half kh selects K block of 16:
//   K = j + 16*kh   (dword-packed pairs)
__device__ __forceinline__ v16bf load_a_frag(const __bf16* __restrict__ base,
                                             int ld, int lane) {
  const int m = lane & 15, kh = (lane >> 4) & 1;
  const uint32_t* r = (const uint32_t*)(base + (size_t)m * ld);
  union { v16bf v; uint32_t d[8]; } u;
#pragma unroll
  for (int i = 0; i < 8; ++i) u.d[i] = r[8 * kh + i];
  return u.v;
}

// B fragment: 32x16 (KxN), source stored column-major in memory as rows of N
// with K contiguous (i.e. src[n][k], row stride ld). Lane n = lane&15,
// lane half selects K block of 16; 16 contiguous K elements per lane.
__device__ __forceinline__ v16bf load_b_frag(const __bf16* __restrict__ base,
                                             int ld, int lane) {
  const int n = lane & 15, kb = (lane >> 4) & 1;
  const uint32_t* r = (const uint32_t*)(base + (size_t)n * ld + 16 * kb);
  union { v16bf v; uint32_t d[8]; } u;
#pragma unroll
  for (int i = 0; i < 8; ++i) u.d[i] = r[i];
  return u.v;
}

// Same B mapping but converting from f32 source (for the tiny dots matrix).
__device__ __forceinline__ v16bf load_b_frag_f32(const float* __restrict__ base,
                                                 int ld, int lane) {
  const int n = lane & 15, kb = (lane >> 4) & 1;
  const float* r = base + (size_t)n * ld + 16 * kb;
  v16bf b;
#pragma unroll
  for (int j = 0; j < 16; ++j) b[j] = (__bf16)r[j];
  return b;
}

// ---------------------------------------------------------------------------
// fp32 -> bf16 convert (grid-stride)
// ---------------------------------------------------------------------------
__global__ void cvt_f32_bf16(const float* __restrict__ in,
                             __bf16* __restrict__ out, long n) {
  long i = (long)blockIdx.x * blockDim.x + threadIdx.x;
  long stride = (long)gridDim.x * blockDim.x;
  for (; i < n; i += stride) out[i] = (__bf16)in[i];
}

// ---------------------------------------------------------------------------
// Generic C = A(MxK) * W(NcolsxK)^T, bf16 WMMA, optional fused 2D-rotary
// epilogue (pairs (d, d+16) live in the same lane of adjacent N-tiles).
// Block = 8 waves, wave tile 32x32, block tile 256x32, grid (M/256, Ncols/32).
// ---------------------------------------------------------------------------
template <bool ROTARY, typename OT>
__global__ __launch_bounds__(256) void gemm_nt(
    const __bf16* __restrict__ A, const __bf16* __restrict__ W,
    OT* __restrict__ C, const float* __restrict__ pos,
    const float* __restrict__ bias, int Ncols, int K) {
  const int lane = threadIdx.x & 31;
  const int wave = threadIdx.x >> 5;
  const int m0 = blockIdx.x * 256 + wave * 32;
  const int n0 = blockIdx.y * 32;

  v8f acc[2][2] = {};
  for (int k0 = 0; k0 < K; k0 += 32) {
    v16bf a0 = load_a_frag(A + (size_t)m0 * K + k0, K, lane);
    v16bf a1 = load_a_frag(A + (size_t)(m0 + 16) * K + k0, K, lane);
    v16bf b0 = load_b_frag(W + (size_t)n0 * K + k0, K, lane);
    v16bf b1 = load_b_frag(W + (size_t)(n0 + 16) * K + k0, K, lane);
    acc[0][0] = wmma_bf16(a0, b0, acc[0][0]);
    acc[0][1] = wmma_bf16(a0, b1, acc[0][1]);
    acc[1][0] = wmma_bf16(a1, b0, acc[1][0]);
    acc[1][1] = wmma_bf16(a1, b1, acc[1][1]);
  }

  if (ROTARY) {
    // wave covers one 32-wide rotary half of a 64-wide head; axis from n0.
    const float invf = exp2f(-(float)(lane & 15) * LOG2_10000_OVER16);
    const int axis = (n0 >> 5) & 1;
#pragma unroll
    for (int mt = 0; mt < 2; ++mt)
#pragma unroll
      for (int r = 0; r < 8; ++r) {
        int m = m0 + mt * 16 + r + 8 * (lane >> 4);
        float th = pos[(size_t)m * 2 + axis] * POS_SCALE * invf;
        float s = sinf(th), c = cosf(th);
        float a0 = acc[mt][0][r], a1 = acc[mt][1][r];
        acc[mt][0][r] = a0 * c - a1 * s;   // d        (rotate_half -> -t[d+16])
        acc[mt][1][r] = a1 * c + a0 * s;   // d + 16   (rotate_half ->  t[d])
      }
  }

#pragma unroll
  for (int mt = 0; mt < 2; ++mt)
#pragma unroll
    for (int nt = 0; nt < 2; ++nt)
#pragma unroll
      for (int r = 0; r < 8; ++r) {
        int m = m0 + mt * 16 + r + 8 * (lane >> 4);
        int n = n0 + nt * 16 + (lane & 15);
        float v = acc[mt][nt][r];
        if (bias) v += bias[n];
        C[(size_t)m * Ncols + n] = (OT)v;
      }
}

// ---------------------------------------------------------------------------
// Instance-norm k,v over head dim (64) + rotary on k; writes kT, vT in
// (b,h,d,n2) layout (n2-contiguous) for coalesced WMMA fragments in dots.
// One wave per (token,head): lane holds d=lane and d=lane+32.
// ---------------------------------------------------------------------------
__global__ __launch_bounds__(256) void norm_rot_kernel(
    const __bf16* __restrict__ kv, const float* __restrict__ zpos,
    __bf16* __restrict__ kT, __bf16* __restrict__ vT) {
  const int lane = threadIdx.x & 31;
  const int wave = threadIdx.x >> 5;
  const int idx = blockIdx.x * 8 + wave;   // row*HEADS + h
  const int row = idx >> 3;
  const int h = idx & 7;

  const __bf16* kp = kv + (size_t)row * (2 * INNER_) + h * 64;
  const __bf16* vp = kp + INNER_;
  float k0 = (float)kp[lane], k1 = (float)kp[lane + 32];
  float v0 = (float)vp[lane], v1 = (float)vp[lane + 32];

  float ks = k0 + k1, kq = k0 * k0 + k1 * k1;
  float vs = v0 + v1, vq = v0 * v0 + v1 * v1;
#pragma unroll
  for (int o = 16; o > 0; o >>= 1) {
    ks += __shfl_xor(ks, o, 32);
    kq += __shfl_xor(kq, o, 32);
    vs += __shfl_xor(vs, o, 32);
    vq += __shfl_xor(vq, o, 32);
  }
  const float inv64 = 1.0f / 64.0f;
  float km = ks * inv64, vm = vs * inv64;
  float kr = rsqrtf(kq * inv64 - km * km + 1e-5f);
  float vr = rsqrtf(vq * inv64 - vm * vm + 1e-5f);
  k0 = (k0 - km) * kr; k1 = (k1 - km) * kr;
  v0 = (v0 - vm) * vr; v1 = (v1 - vm) * vr;

  // 2D rotary on normalized k: d=lane uses axis x, d=lane+32 uses axis y.
  float px = zpos[(size_t)row * 2 + 0];
  float py = zpos[(size_t)row * 2 + 1];
  float invf = exp2f(-(float)(lane & 15) * LOG2_10000_OVER16);
  float th0 = px * POS_SCALE * invf;
  float th1 = py * POS_SCALE * invf;
  float p0 = __shfl_xor(k0, 16, 32);
  float p1 = __shfl_xor(k1, 16, 32);
  float sgn = (lane < 16) ? -1.0f : 1.0f;
  float k0r = k0 * cosf(th0) + sgn * p0 * sinf(th0);
  float k1r = k1 * cosf(th1) + sgn * p1 * sinf(th1);

  const int b = row >> 13;            // row / N2
  const int n2 = row & (N2_ - 1);
  const size_t bh = (size_t)b * HEADS_ + h;
  __bf16* kd = kT + (bh * 64) * (size_t)N2_ + n2;
  __bf16* vd = vT + (bh * 64) * (size_t)N2_ + n2;
  kd[(size_t)lane * N2_]        = (__bf16)k0r;
  kd[(size_t)(lane + 32) * N2_] = (__bf16)k1r;
  vd[(size_t)lane * N2_]        = (__bf16)v0;
  vd[(size_t)(lane + 32) * N2_] = (__bf16)v1;
}

// ---------------------------------------------------------------------------
// dotsT[bh][e][d] += sum_n2 k[n2][d] * v[n2][e]  (stored e-major for q@dots)
// grid (B*H, KSPLIT); 8 waves each reduce a K chunk into full 64x64 tile,
// LDS ds_add_f32 combine, then global f32 atomics (dots is L2-resident).
// ---------------------------------------------------------------------------
__global__ __launch_bounds__(256) void dots_kernel(
    const __bf16* __restrict__ kT, const __bf16* __restrict__ vT,
    float* __restrict__ dotsT) {
  __shared__ float red[64 * 64];
  const int lane = threadIdx.x & 31;
  const int wave = threadIdx.x >> 5;
  const int bh = blockIdx.x;
  const int KC = N2_ / KSPLIT;                 // 1024 per block
  const int kbase = blockIdx.y * KC + wave * (KC / 8);   // 128 per wave
  const __bf16* Ab = kT + (size_t)bh * 64 * N2_;
  const __bf16* Bb = vT + (size_t)bh * 64 * N2_;

  v8f acc[4][4] = {};
  for (int k0 = kbase; k0 < kbase + KC / 8; k0 += 32) {
    v16bf a[4], bb[4];
#pragma unroll
    for (int i = 0; i < 4; ++i)
      a[i] = load_a_frag(Ab + (size_t)(i * 16) * N2_ + k0, N2_, lane);
#pragma unroll
    for (int i = 0; i < 4; ++i)
      bb[i] = load_b_frag(Bb + (size_t)(i * 16) * N2_ + k0, N2_, lane);
#pragma unroll
    for (int mt = 0; mt < 4; ++mt)
#pragma unroll
      for (int nt = 0; nt < 4; ++nt)
        acc[mt][nt] = wmma_bf16(a[mt], bb[nt], acc[mt][nt]);
  }

  for (int i = threadIdx.x; i < 4096; i += 256) red[i] = 0.0f;
  __syncthreads();
#pragma unroll
  for (int mt = 0; mt < 4; ++mt)
#pragma unroll
    for (int nt = 0; nt < 4; ++nt)
#pragma unroll
      for (int r = 0; r < 8; ++r) {
        int m = mt * 16 + r + 8 * (lane >> 4);   // d
        int n = nt * 16 + (lane & 15);           // e
        atomicAdd(&red[n * 64 + m], acc[mt][nt][r]);
      }
  __syncthreads();
  float* dst = dotsT + (size_t)bh * 4096;
  for (int i = threadIdx.x; i < 4096; i += 256) atomicAdd(&dst[i], red[i]);
}

// ---------------------------------------------------------------------------
// attn[m][h*64+e] = (1/N2) * sum_d q[m][h*64+d] * dotsT[bh][e][d]
// grid (MTOT/256, HEADS); wave tile 32(M) x 64(N=e), K=64.
// ---------------------------------------------------------------------------
__global__ __launch_bounds__(256) void attn_kernel(
    const __bf16* __restrict__ q, const float* __restrict__ dotsT,
    __bf16* __restrict__ attn) {
  const int lane = threadIdx.x & 31;
  const int wave = threadIdx.x >> 5;
  const int m0 = blockIdx.x * 256 + wave * 32;
  const int h = blockIdx.y;
  const int b = m0 >> 13;
  const float* D = dotsT + ((size_t)(b * HEADS_ + h)) * 4096;

  v8f acc[2][4] = {};
#pragma unroll
  for (int k0 = 0; k0 < 64; k0 += 32) {
    v16bf a0 = load_a_frag(q + (size_t)m0 * INNER_ + h * 64 + k0, INNER_, lane);
    v16bf a1 = load_a_frag(q + (size_t)(m0 + 16) * INNER_ + h * 64 + k0, INNER_, lane);
    v16bf bb[4];
#pragma unroll
    for (int nt = 0; nt < 4; ++nt)
      bb[nt] = load_b_frag_f32(D + nt * 16 * 64 + k0, 64, lane);
#pragma unroll
    for (int nt = 0; nt < 4; ++nt) {
      acc[0][nt] = wmma_bf16(a0, bb[nt], acc[0][nt]);
      acc[1][nt] = wmma_bf16(a1, bb[nt], acc[1][nt]);
    }
  }
  const float scale = 1.0f / (float)N2_;
#pragma unroll
  for (int mt = 0; mt < 2; ++mt)
#pragma unroll
    for (int nt = 0; nt < 4; ++nt)
#pragma unroll
      for (int r = 0; r < 8; ++r) {
        int m = m0 + mt * 16 + r + 8 * (lane >> 4);
        int n = h * 64 + nt * 16 + (lane & 15);
        attn[(size_t)m * INNER_ + n] = (__bf16)(acc[mt][nt][r] * scale);
      }
}

// ---------------------------------------------------------------------------
// Host orchestration
// ---------------------------------------------------------------------------
extern "C" void kernel_launch(void* const* d_in, const int* in_sizes, int n_in,
                              void* d_out, int out_size, void* d_ws, size_t ws_size,
                              hipStream_t stream) {
  (void)in_sizes; (void)n_in; (void)out_size; (void)ws_size;
  const float* x    = (const float*)d_in[0];
  const float* z    = (const float*)d_in[1];
  const float* xpos = (const float*)d_in[2];
  const float* zpos = (const float*)d_in[3];
  const float* Wq   = (const float*)d_in[4];
  const float* Wkv  = (const float*)d_in[5];
  const float* Wout = (const float*)d_in[6];
  const float* bout = (const float*)d_in[7];
  float* out = (float*)d_out;

  char* p = (char*)d_ws;
  auto carve = [&](size_t bytes) {
    char* r = p; p += (bytes + 255) & ~(size_t)255; return r;
  };
  __bf16* xb    = (__bf16*)carve((size_t)MTOT * DIM_ * 2);
  __bf16* zb    = (__bf16*)carve((size_t)MTOT * DIM_ * 2);
  __bf16* wqb   = (__bf16*)carve((size_t)INNER_ * DIM_ * 2);
  __bf16* wkvb  = (__bf16*)carve((size_t)2 * INNER_ * DIM_ * 2);
  __bf16* wob   = (__bf16*)carve((size_t)DIM_ * INNER_ * 2);
  __bf16* qb    = (__bf16*)carve((size_t)MTOT * INNER_ * 2);
  __bf16* kvb   = (__bf16*)carve((size_t)MTOT * 2 * INNER_ * 2);
  __bf16* kT    = (__bf16*)carve((size_t)B_ * HEADS_ * DH_ * N2_ * 2);
  __bf16* vT    = (__bf16*)carve((size_t)B_ * HEADS_ * DH_ * N2_ * 2);
  float*  dotsT = (float*)carve((size_t)B_ * HEADS_ * DH_ * DH_ * 4);
  __bf16* attnb = kvb;   // kv is dead after norm_rot; reuse for attn output

  auto cvt = [&](const float* s, __bf16* d, long n) {
    int blk = (int)((n + 2047) / 2048);
    if (blk > 4096) blk = 4096;
    cvt_f32_bf16<<<dim3(blk), dim3(256), 0, stream>>>(s, d, n);
  };
  cvt(x,    xb,   (long)MTOT * DIM_);
  cvt(z,    zb,   (long)MTOT * DIM_);
  cvt(Wq,   wqb,  (long)INNER_ * DIM_);
  cvt(Wkv,  wkvb, (long)2 * INNER_ * DIM_);
  cvt(Wout, wob,  (long)DIM_ * INNER_);

  // q = x @ Wq^T with fused rotary
  gemm_nt<true, __bf16><<<dim3(MTOT / 256, INNER_ / 32), dim3(256), 0, stream>>>(
      xb, wqb, qb, xpos, nullptr, INNER_, DIM_);

  // kv = z @ Wkv^T
  gemm_nt<false, __bf16><<<dim3(MTOT / 256, (2 * INNER_) / 32), dim3(256), 0, stream>>>(
      zb, wkvb, kvb, nullptr, nullptr, 2 * INNER_, DIM_);

  // instance norm + rotary(k) -> transposed kT/vT
  norm_rot_kernel<<<dim3(MTOT * HEADS_ / 8), dim3(256), 0, stream>>>(
      kvb, zpos, kT, vT);

  // dots = k^T v, K-split reduction
  hipMemsetAsync(dotsT, 0, (size_t)B_ * HEADS_ * DH_ * DH_ * 4, stream);
  dots_kernel<<<dim3(B_ * HEADS_, KSPLIT), dim3(256), 0, stream>>>(kT, vT, dotsT);

  // attn = (q @ dots) / N2
  attn_kernel<<<dim3(MTOT / 256, HEADS_), dim3(256), 0, stream>>>(qb, dotsT, attnb);

  // out = attn @ Wout^T + bout
  gemm_nt<false, float><<<dim3(MTOT / 256, DIM_ / 32), dim3(256), 0, stream>>>(
      attnb, wob, out, nullptr, bout, DIM_, INNER_);
}
